// Attention_87557203296748
// MI455X (gfx1250) — compile-verified, matrix operand order checked by reading, not run on hardware
//
#include <hip/hip_runtime.h>
#include <hip/hip_bf16.h>
#include <math.h>

// ---------------------------------------------------------------------------
// MHA forward for B=4, N=2048, D=1024, H=16, HD=64 on gfx1250 (wave32, WMMA).
// bf16 inputs to v_wmma_f32_16x16x32_bf16, fp32 accumulation throughout.
// Compute-bound (~137 GFLOP vs <10us of HBM traffic at 23.3 TB/s), so all
// three stages run through the matrix pipes.
// ---------------------------------------------------------------------------

typedef __bf16 bf16_t;
typedef __attribute__((ext_vector_type(16))) __bf16 v16bf;
typedef __attribute__((ext_vector_type(8)))  __bf16 v8bf;
typedef __attribute__((ext_vector_type(8)))  float  v8f;

#define BATCH 4
#define SEQ   2048
#define DMODEL 1024
#define NHEAD 16
#define HDIM  64
#define BN    (BATCH * SEQ)       // 8192
#define E3    (3 * DMODEL)        // 3072
#define SCALE 0.125f              // 64^-0.5

__device__ __forceinline__ v8f wmma_bf16(v16bf a, v16bf b, v8f c) {
  return __builtin_amdgcn_wmma_f32_16x16x32_bf16(
      /*neg_a=*/false, a, /*neg_b=*/false, b,
      /*c_mod=*/(short)0, c, /*reuse_a=*/false, /*reuse_b=*/false);
}

// Load one 16x32 bf16 fragment (A-layout; identical addressing serves the
// B-fragment of a TN gemm where B rows are K-contiguous).
//   lane 0-15 : row = lane,    k = {0..7, 16..23}
//   lane 16-31: row = lane-16, k = {8..15, 24..31}
// Two 16-byte reads per lane; `t` may point at LDS (ds_load after addrspace
// inference) or global memory.
__device__ __forceinline__ v16bf ld_frag(const bf16_t* t, int row, int stride,
                                         int kbase) {
  const int lane = threadIdx.x & 31;
  const int r = lane & 15;
  const int klo = (lane >> 4) << 3;
  const bf16_t* p = t + (size_t)(row + r) * stride + kbase + klo;
  union { v16bf v; v8bf h[2]; } u;
  u.h[0] = *(const v8bf*)(p);
  u.h[1] = *(const v8bf*)(p + 16);
  return u.v;
}

// ---------------------------------------------------------------------------
// Kernel 1: qkv = x @ w_qkv^T  (TN gemm, M=8192, N=3072, K=1024)
// WG tile 128x128, 8 waves, each wave 32x64 (acc[2][4], 8 WMMAs / K-step).
// Scatters to q (pre-scaled), k as (B,H,N,HD) and v transposed (B,H,HD,N).
// ---------------------------------------------------------------------------
__global__ __launch_bounds__(256) void qkv_gemm_kernel(
    const float* __restrict__ x, const float* __restrict__ w_qkv,
    bf16_t* __restrict__ qw, bf16_t* __restrict__ kw, bf16_t* __restrict__ vtw) {
  __shared__ bf16_t Xs[128 * 40];   // 128 rows x 32 K, stride 40 (pad)
  __shared__ bf16_t Ws[128 * 40];   // 128 rows x 32 K

  const int tid  = threadIdx.x;
  const int wave = tid >> 5;
  const int wm   = wave >> 1;       // 0..3 -> 32-row block
  const int wn   = wave & 1;        // 0..1 -> 64-col block
  const int m0   = blockIdx.x * 128;
  const int e0   = blockIdx.y * 128;

  v8f acc[2][4];
  for (int i = 0; i < 2; ++i)
    for (int j = 0; j < 4; ++j) acc[i][j] = 0.f;

  for (int kb = 0; kb < DMODEL; kb += 32) {
    __syncthreads();
    // Stage X tile 128x32 fp32 -> bf16 LDS (1024 float4, 4 per thread).
    for (int i = 0; i < 4; ++i) {
      int gid = tid + i * 256;
      int r = gid >> 3, c = (gid & 7) * 4;
      const float4 f =
          *(const float4*)(x + (size_t)(m0 + r) * DMODEL + kb + c);
      union { bf16_t b[4]; uint2 u; } pk;
      pk.b[0] = (bf16_t)f.x; pk.b[1] = (bf16_t)f.y;
      pk.b[2] = (bf16_t)f.z; pk.b[3] = (bf16_t)f.w;
      *(uint2*)&Xs[r * 40 + c] = pk.u;
    }
    // Stage W tile 128x32 fp32 -> bf16 LDS.
    for (int i = 0; i < 4; ++i) {
      int gid = tid + i * 256;
      int r = gid >> 3, c = (gid & 7) * 4;
      const float4 f =
          *(const float4*)(w_qkv + (size_t)(e0 + r) * DMODEL + kb + c);
      union { bf16_t b[4]; uint2 u; } pk;
      pk.b[0] = (bf16_t)f.x; pk.b[1] = (bf16_t)f.y;
      pk.b[2] = (bf16_t)f.z; pk.b[3] = (bf16_t)f.w;
      *(uint2*)&Ws[r * 40 + c] = pk.u;
    }
    __syncthreads();

    v16bf a0 = ld_frag(Xs, wm * 32,      40, 0);
    v16bf a1 = ld_frag(Xs, wm * 32 + 16, 40, 0);
    for (int j = 0; j < 4; ++j) {
      v16bf bj = ld_frag(Ws, wn * 64 + j * 16, 40, 0);
      acc[0][j] = wmma_bf16(a0, bj, acc[0][j]);
      acc[1][j] = wmma_bf16(a1, bj, acc[1][j]);
    }
  }

  // Epilogue scatter. C-layout: lane holds col=lane&15, rows v+8*(lane>>4).
  // The 128-wide N tile lies inside one q/k/v segment -> s is block-uniform.
  const int lane = tid & 31;
  const int half = lane >> 4;
  const int col  = lane & 15;
  const int s    = e0 >> 10;                        // 0=q, 1=k, 2=v (uniform)
  const int gm0  = m0 + wm * 32;
  const int ge0  = e0 + wn * 64;

  if (s == 0) {
    for (int mi = 0; mi < 2; ++mi)
      for (int ni = 0; ni < 4; ++ni)
        for (int v = 0; v < 8; ++v) {
          int gm = gm0 + mi * 16 + v + 8 * half;
          int ge = ge0 + ni * 16 + col;
          int b = gm >> 11, n = gm & (SEQ - 1);
          int h = (ge >> 6) & (NHEAD - 1), hd = ge & (HDIM - 1);
          qw[((size_t)(b * NHEAD + h) * SEQ + n) * HDIM + hd] =
              (bf16_t)(acc[mi][ni][v] * SCALE);
        }
  } else if (s == 1) {
    for (int mi = 0; mi < 2; ++mi)
      for (int ni = 0; ni < 4; ++ni)
        for (int v = 0; v < 8; ++v) {
          int gm = gm0 + mi * 16 + v + 8 * half;
          int ge = ge0 + ni * 16 + col;
          int b = gm >> 11, n = gm & (SEQ - 1);
          int h = (ge >> 6) & (NHEAD - 1), hd = ge & (HDIM - 1);
          kw[((size_t)(b * NHEAD + h) * SEQ + n) * HDIM + hd] =
              (bf16_t)acc[mi][ni][v];
        }
  } else {
    for (int mi = 0; mi < 2; ++mi)
      for (int ni = 0; ni < 4; ++ni)
        for (int v = 0; v < 8; ++v) {
          int gm = gm0 + mi * 16 + v + 8 * half;
          int ge = ge0 + ni * 16 + col;
          int b = gm >> 11, n = gm & (SEQ - 1);
          int h = (ge >> 6) & (NHEAD - 1), hd = ge & (HDIM - 1);
          vtw[((size_t)(b * NHEAD + h) * HDIM + hd) * SEQ + n] =
              (bf16_t)acc[mi][ni][v];
        }
  }
}

// ---------------------------------------------------------------------------
// Kernel 2: flash attention. One WG (128 thr = 4 waves) per 64 q-rows of one
// (b,h). Each wave owns 16 q-rows; KV streamed in 64-wide tiles via LDS.
// ---------------------------------------------------------------------------
__global__ __launch_bounds__(128) void attn_kernel(
    const bf16_t* __restrict__ qw, const bf16_t* __restrict__ kw,
    const bf16_t* __restrict__ vtw, bf16_t* __restrict__ ow) {
  __shared__ bf16_t Ks[64 * 72];     // K tile: rows=kv, cols=hd
  __shared__ bf16_t Vs[64 * 72];     // V^T tile: rows=hd, cols=kv
  __shared__ bf16_t Ps[4][16 * 72];  // per-wave P scratch (C->A relayout)

  const int tid  = threadIdx.x;
  const int wave = tid >> 5;
  const int lane = tid & 31;
  const int half = lane >> 4;
  const int col  = lane & 15;
  const int bh   = blockIdx.y;                 // b*16 + h
  const int qrow0 = blockIdx.x * 64 + wave * 16;

  // Q fragments (16 rows x HD=64 -> two k-slices), straight from global.
  const bf16_t* qbase = qw + ((size_t)bh * SEQ + qrow0) * HDIM;
  v16bf qf[2];
  qf[0] = ld_frag(qbase, 0, HDIM, 0);
  qf[1] = ld_frag(qbase, 0, HDIM, 32);

  v8f of[4];
  of[0] = 0.f; of[1] = 0.f; of[2] = 0.f; of[3] = 0.f;
  float m_i[8], l_i[8];
  for (int v = 0; v < 8; ++v) { m_i[v] = -1e30f; l_i[v] = 0.f; }

  const bf16_t* kg0 = kw + (size_t)bh * SEQ * HDIM;
  const bf16_t* vg0 = vtw + (size_t)bh * HDIM * SEQ;

  for (int kv0 = 0; kv0 < SEQ; kv0 += 64) {
    __syncthreads();
    // Stage K tile (64 kv-rows x 64 hd) and V^T tile (64 hd-rows x 64 kv).
    for (int i = 0; i < 4; ++i) {
      int gid = tid + i * 128;
      int r = gid >> 3, c = (gid & 7) * 8;
      *(uint4*)&Ks[r * 72 + c] =
          *(const uint4*)(kg0 + (size_t)(kv0 + r) * HDIM + c);
    }
    for (int i = 0; i < 4; ++i) {
      int gid = tid + i * 128;
      int r = gid >> 3, c = (gid & 7) * 8;
      *(uint4*)&Vs[r * 72 + c] =
          *(const uint4*)(vg0 + (size_t)r * SEQ + kv0 + c);
    }
    __syncthreads();

    // S = Q K^T : 4 column blocks, K-dim 64 -> 2 WMMAs each.
    v8f sf[4];
    for (int j = 0; j < 4; ++j) {
      v8f s = 0.f;
      s = wmma_bf16(qf[0], ld_frag(Ks, j * 16, 72, 0),  s);
      s = wmma_bf16(qf[1], ld_frag(Ks, j * 16, 72, 32), s);
      sf[j] = s;
    }

    // Online softmax. Row M = v + 8*half lives across a 16-lane half-wave.
    float mt[8], scl[8], rs[8];
    for (int v = 0; v < 8; ++v) {
      float m = sf[0][v];
      m = fmaxf(m, sf[1][v]); m = fmaxf(m, sf[2][v]); m = fmaxf(m, sf[3][v]);
      mt[v] = m;
    }
    for (int off = 1; off < 16; off <<= 1)
      for (int v = 0; v < 8; ++v)
        mt[v] = fmaxf(mt[v], __shfl_xor(mt[v], off, 32));
    for (int v = 0; v < 8; ++v) {
      float mn = fmaxf(m_i[v], mt[v]);
      scl[v] = __expf(m_i[v] - mn);
      m_i[v] = mn;
    }
    for (int v = 0; v < 8; ++v) {
      float sum = 0.f;
      for (int j = 0; j < 4; ++j) {
        float p = __expf(sf[j][v] - m_i[v]);
        sf[j][v] = p;
        sum += p;
      }
      rs[v] = sum;
    }
    for (int off = 1; off < 16; off <<= 1)
      for (int v = 0; v < 8; ++v)
        rs[v] += __shfl_xor(rs[v], off, 32);
    for (int v = 0; v < 8; ++v) l_i[v] = l_i[v] * scl[v] + rs[v];
    for (int j = 0; j < 4; ++j)
      for (int v = 0; v < 8; ++v) of[j][v] *= scl[v];

    // P: C-fragment layout -> per-wave LDS -> A-fragment layout.
    // Wave-private region; DS ops within a wave are in-order (RAW safe).
    bf16_t* pw = &Ps[wave][0];
    for (int j = 0; j < 4; ++j)
      for (int v = 0; v < 8; ++v)
        pw[(v + 8 * half) * 72 + j * 16 + col] = (bf16_t)sf[j][v];

    // O += P V  (K-dim 64 over kv, 4 hd column blocks).
    v16bf p0 = ld_frag(pw, 0, 72, 0);
    v16bf p1 = ld_frag(pw, 0, 72, 32);
    for (int j = 0; j < 4; ++j) {
      of[j] = wmma_bf16(p0, ld_frag(Vs, j * 16, 72, 0),  of[j]);
      of[j] = wmma_bf16(p1, ld_frag(Vs, j * 16, 72, 32), of[j]);
    }
  }

  // Finalize: O *= 1/l, write to (B, N, H*HD) bf16 for the projection gemm.
  const int b = bh >> 4, h = bh & 15;
  float inv_l[8];
  for (int v = 0; v < 8; ++v) inv_l[v] = 1.0f / l_i[v];
  for (int j = 0; j < 4; ++j)
    for (int v = 0; v < 8; ++v) {
      int n = qrow0 + v + 8 * half;
      float val = of[j][v] * inv_l[v];
      ow[((size_t)(b * SEQ + n)) * DMODEL + h * HDIM + j * 16 + col] =
          (bf16_t)val;
    }
}

// ---------------------------------------------------------------------------
// Kernel 3: out = attn_out @ w_proj^T + b_proj  (M=8192, N=1024, K=1024)
// WG tile 128x128, 8 waves, each wave 32x64.
// ---------------------------------------------------------------------------
__global__ __launch_bounds__(256) void proj_gemm_kernel(
    const bf16_t* __restrict__ a, const float* __restrict__ w_proj,
    const float* __restrict__ b_proj, float* __restrict__ out) {
  __shared__ bf16_t As[128 * 40];
  __shared__ bf16_t Ws[128 * 40];

  const int tid  = threadIdx.x;
  const int wave = tid >> 5;
  const int wm   = wave >> 1;
  const int wn   = wave & 1;
  const int m0   = blockIdx.x * 128;
  const int e0   = blockIdx.y * 128;

  v8f acc[2][4];
  for (int i = 0; i < 2; ++i)
    for (int j = 0; j < 4; ++j) acc[i][j] = 0.f;

  for (int kb = 0; kb < DMODEL; kb += 32) {
    __syncthreads();
    // Stage A tile 128x32 bf16 (512 uint4, 2 per thread).
    for (int i = 0; i < 2; ++i) {
      int gid = tid + i * 256;
      int r = gid >> 2, c = (gid & 3) * 8;
      *(uint4*)&As[r * 40 + c] =
          *(const uint4*)(a + (size_t)(m0 + r) * DMODEL + kb + c);
    }
    // Stage W tile 128x32 fp32 -> bf16 (1024 float4, 4 per thread).
    for (int i = 0; i < 4; ++i) {
      int gid = tid + i * 256;
      int r = gid >> 3, c = (gid & 7) * 4;
      const float4 f =
          *(const float4*)(w_proj + (size_t)(e0 + r) * DMODEL + kb + c);
      union { bf16_t b[4]; uint2 u; } pk;
      pk.b[0] = (bf16_t)f.x; pk.b[1] = (bf16_t)f.y;
      pk.b[2] = (bf16_t)f.z; pk.b[3] = (bf16_t)f.w;
      *(uint2*)&Ws[r * 40 + c] = pk.u;
    }
    __syncthreads();

    v16bf a0 = ld_frag(As, wm * 32,      40, 0);
    v16bf a1 = ld_frag(As, wm * 32 + 16, 40, 0);
    for (int j = 0; j < 4; ++j) {
      v16bf bj = ld_frag(Ws, wn * 64 + j * 16, 40, 0);
      acc[0][j] = wmma_bf16(a0, bj, acc[0][j]);
      acc[1][j] = wmma_bf16(a1, bj, acc[1][j]);
    }
  }

  const int lane = tid & 31;
  const int half = lane >> 4;
  const int col  = lane & 15;
  for (int mi = 0; mi < 2; ++mi)
    for (int ni = 0; ni < 4; ++ni)
      for (int v = 0; v < 8; ++v) {
        int gm = m0 + wm * 32 + mi * 16 + v + 8 * half;
        int ge = e0 + wn * 64 + ni * 16 + col;
        out[(size_t)gm * DMODEL + ge] = acc[mi][ni][v] + b_proj[ge];
      }
}

// ---------------------------------------------------------------------------
extern "C" void kernel_launch(void* const* d_in, const int* in_sizes, int n_in,
                              void* d_out, int out_size, void* d_ws,
                              size_t ws_size, hipStream_t stream) {
  const float* x      = (const float*)d_in[0];
  const float* w_qkv  = (const float*)d_in[1];
  const float* w_proj = (const float*)d_in[2];
  const float* b_proj = (const float*)d_in[3];
  float* out = (float*)d_out;

  const size_t PER = (size_t)BATCH * NHEAD * SEQ * HDIM;  // 8M elements
  bf16_t* qw  = (bf16_t*)d_ws;
  bf16_t* kw  = qw + PER;
  bf16_t* vtw = kw + PER;
  bf16_t* ow  = vtw + PER;   // (B, N, D) attention output, bf16

  qkv_gemm_kernel<<<dim3(BN / 128, E3 / 128), 256, 0, stream>>>(
      x, w_qkv, qw, kw, vtw);
  attn_kernel<<<dim3(SEQ / 64, BATCH * NHEAD), 128, 0, stream>>>(
      qw, kw, vtw, ow);
  proj_gemm_kernel<<<dim3(BN / 128, DMODEL / 128), 256, 0, stream>>>(
      ow, w_proj, b_proj, out);
}
